// STAttnBlock_28097676051224
// MI455X (gfx1250) — matrix operand order repaired and checked by reading, not compiled, not run on hardware
//
#include <hip/hip_runtime.h>

#define BATCH 2
#define CCH   256
#define FF    3
#define HWSZ  2304            // 48*48
#define NPOS  6912            // FF*HWSZ
#define NGRP  4
#define CG    64              // CCH/NGRP

typedef __attribute__((ext_vector_type(16))) __bf16        bf16x16;
typedef __attribute__((ext_vector_type(8)))  float         f32x8;
typedef __attribute__((ext_vector_type(8)))  unsigned int  u32x8;

static __device__ __forceinline__ unsigned short f2bf(float f) {
    unsigned int u = __float_as_uint(f);
    u = (u + 0x7FFFu + ((u >> 16) & 1u)) >> 16;   // round-to-nearest-even
    return (unsigned short)u;
}

static __device__ __forceinline__ f32x8 wmma_bf16(const u32x8& a, const u32x8& b, f32x8 c) {
    return __builtin_amdgcn_wmma_f32_16x16x32_bf16(
        false, __builtin_bit_cast(bf16x16, a),
        false, __builtin_bit_cast(bf16x16, b),
        (short)0, c, false, false);
}

// A-fragment (16x32 bf16, MxK): per-lane K-pair offset for VGPR r
static __device__ __forceinline__ int akb(int r, int half) {
    return ((r >> 2) << 4) + ((r & 3) << 1) + (half << 3);
}
// B-fragment (32x16 bf16, KxN): per-lane K-pair offset for VGPR r
static __device__ __forceinline__ int bkb(int r, int half) {
    return ((r >> 2) << 4) + ((r & 3) << 1) + (half << 3);
}

// ---------------- Kernel 1: GroupNorm statistics -------------------------
__global__ __launch_bounds__(256) void gn_stats(const float* __restrict__ x,
                                                float* __restrict__ meanrstd) {
    int bg = blockIdx.x;                 // b*4 + g
    int b = bg >> 2, g = bg & 3;
    int tid = threadIdx.x;
    float s = 0.f, ss = 0.f;
    const int total = CG * FF * HWSZ;    // 442368
    for (int idx = tid; idx < total; idx += 256) {
        int p  = idx % HWSZ;
        int t  = idx / HWSZ;
        int f  = t % FF;
        int ch = g * CG + t / FF;
        float v = x[((size_t)(f * 2 + b) * CCH + ch) * HWSZ + p];
        s += v; ss += v * v;
    }
    __shared__ float sh[512];
    sh[tid] = s; sh[256 + tid] = ss;
    __syncthreads();
    for (int st = 128; st > 0; st >>= 1) {
        if (tid < st) { sh[tid] += sh[tid + st]; sh[256 + tid] += sh[256 + tid + st]; }
        __syncthreads();
    }
    if (tid == 0) {
        float inv = 1.0f / (float)total;
        float m   = sh[0] * inv;
        float var = sh[256] * inv - m * m;
        meanrstd[bg * 2 + 0] = m;
        meanrstd[bg * 2 + 1] = rsqrtf(var + 1e-6f);
    }
}

// ---- Kernel 2: normalize -> bf16 (pos, ch) via LDS tile transpose -------
__global__ __launch_bounds__(256) void gn_apply(const float* __restrict__ x,
                                                const float* __restrict__ meanrstd,
                                                const float* __restrict__ gsc,
                                                const float* __restrict__ gbi,
                                                unsigned short* __restrict__ hn) {
    int p0 = blockIdx.x * 64;
    int c0 = blockIdx.y * 64;            // aligned with 64-channel groups
    int bf = blockIdx.z;                 // b*3 + f
    int b  = bf / 3, f = bf % 3;
    __shared__ float tile[64][65];
    int tid  = threadIdx.x;
    int col  = tid & 63, row4 = tid >> 6;
    int g    = c0 >> 6;
    float m  = meanrstd[(b * 4 + g) * 2 + 0];
    float rs = meanrstd[(b * 4 + g) * 2 + 1];
    for (int cy = row4; cy < 64; cy += 4) {       // coalesced read over pixels
        int ch = c0 + cy;
        float v = x[((size_t)(f * 2 + b) * CCH + ch) * HWSZ + p0 + col];
        tile[cy][col] = (v - m) * rs * gsc[ch] + gbi[ch];
    }
    __syncthreads();
    for (int py = row4; py < 64; py += 4) {       // coalesced write over channels
        int pos = f * HWSZ + p0 + py;
        hn[((size_t)b * NPOS + pos) * CCH + c0 + col] = f2bf(tile[col][py]);
    }
}

// ---------------- Kernel 3: weight fp32 -> bf16 --------------------------
__global__ __launch_bounds__(256) void convert_w(const float* __restrict__ w,
                                                 unsigned short* __restrict__ o, int n) {
    int i = blockIdx.x * 256 + threadIdx.x;
    if (i < n) o[i] = f2bf(w[i]);
}

// ---------------- Kernel 4: QKV projection GEMM (WMMA) -------------------
// trans==0 -> A=hn(pos,K=ch), B=W^T(n=o) -> store (pos, ch) coalesced  [q,k]
// trans==1 -> A=W(o,K=ch),   B=hn(n=pos) -> store (ch, pos) coalesced  [v]
__global__ __launch_bounds__(256) void qkv_gemm(const unsigned short* __restrict__ hn,
                                                const unsigned short* __restrict__ wt,
                                                const float* __restrict__ bias,
                                                unsigned short* __restrict__ out,
                                                int trans) {
    int b  = blockIdx.y;
    int n0 = blockIdx.x * 64;
    __shared__ unsigned short sh[64 * 264];     // 64 pos x 256 ch, padded stride
    for (int i = threadIdx.x; i < 64 * 64; i += 256) {
        int row = i >> 6, c4 = (i & 63) * 4;
        *(uint2*)&sh[row * 264 + c4] =
            *(const uint2*)&hn[((size_t)(b * NPOS + n0 + row)) * CCH + c4];
    }
    __syncthreads();
    int lane = threadIdx.x & 31, wave = threadIdx.x >> 5;
    int half = lane >> 4, l15 = lane & 15;
    for (int t = wave * 8; t < wave * 8 + 8; ++t) {
        int ot = t >> 2, nt = t & 3;            // 16 o-tiles x 4 pos-tiles
        int o0 = ot * 16, prow = nt * 16;
        f32x8 acc = {0.f, 0.f, 0.f, 0.f, 0.f, 0.f, 0.f, 0.f};
        if (trans == 0) {
            for (int kt = 0; kt < 8; ++kt) {
                int k0 = kt * 32;
                u32x8 a, bb;
#pragma unroll
                for (int r = 0; r < 8; ++r) {
                    a[r]  = *(const unsigned int*)&sh[(prow + l15) * 264 + k0 + akb(r, half)];
                    bb[r] = *(const unsigned int*)&wt[(o0 + l15) * CCH + k0 + bkb(r, half)];
                }
                acc = wmma_bf16(a, bb, acc);
            }
            float bcol = bias[o0 + l15];
#pragma unroll
            for (int r = 0; r < 8; ++r) {
                int pos = n0 + prow + r + (half << 3);
                out[((size_t)b * NPOS + pos) * CCH + o0 + l15] = f2bf(acc[r] + bcol);
            }
        } else {
            for (int kt = 0; kt < 8; ++kt) {
                int k0 = kt * 32;
                u32x8 a, bb;
#pragma unroll
                for (int r = 0; r < 8; ++r) {
                    a[r]  = *(const unsigned int*)&wt[(o0 + l15) * CCH + k0 + akb(r, half)];
                    bb[r] = *(const unsigned int*)&sh[(prow + l15) * 264 + k0 + bkb(r, half)];
                }
                acc = wmma_bf16(a, bb, acc);
            }
#pragma unroll
            for (int r = 0; r < 8; ++r) {
                int o   = o0 + r + (half << 3);
                int pos = n0 + prow + l15;
                out[((size_t)b * CCH + o) * NPOS + pos] = f2bf(acc[r] + bias[o]);
            }
        }
    }
}

// ------ Kernel 5: single-pass FlashAttention-2 (QK^T, softmax, @V) -------
// Per wave: 16 query rows; O accum 16x256 fp32 (16 v8f tiles).
// Online max/sum; O rescale gated behind wave-uniform vote.
__global__ __launch_bounds__(256) void flash_attn(const unsigned short* __restrict__ q,
                                                  const unsigned short* __restrict__ k,
                                                  const unsigned short* __restrict__ v,
                                                  unsigned short* __restrict__ hat) {
    int b    = blockIdx.y;
    int wave = threadIdx.x >> 5;
    int q0   = blockIdx.x * 128 + wave * 16;
    int lane = threadIdx.x & 31, half = lane >> 4, l15 = lane & 15;
    const float sc = 0.0625f;   // 256^-0.5

    __shared__ unsigned short pls[8][16 * 34];   // per-wave P re-layout scratch

    float m8[8], l8[8];
#pragma unroll
    for (int r = 0; r < 8; ++r) { m8[r] = -1e30f; l8[r] = 0.f; }

    f32x8 o[16];
#pragma unroll
    for (int ct = 0; ct < 16; ++ct) o[ct] = (f32x8){0.f,0.f,0.f,0.f,0.f,0.f,0.f,0.f};

    const size_t qrow = ((size_t)b * NPOS + q0 + l15) * CCH;

    for (int j0 = 0; j0 < NPOS; j0 += 32) {
        // two 16x16 S tiles (keys j0..j0+31)
        f32x8 s0 = {0.f,0.f,0.f,0.f,0.f,0.f,0.f,0.f};
        f32x8 s1 = {0.f,0.f,0.f,0.f,0.f,0.f,0.f,0.f};
        for (int kt = 0; kt < 8; ++kt) {
            u32x8 a, bb;
#pragma unroll
            for (int r = 0; r < 8; ++r) {
                a[r]  = *(const unsigned int*)&q[qrow + kt * 32 + akb(r, half)];
                bb[r] = *(const unsigned int*)&k[((size_t)b * NPOS + j0 + l15) * CCH +
                                                 kt * 32 + bkb(r, half)];
            }
            s0 = wmma_bf16(a, bb, s0);
        }
        for (int kt = 0; kt < 8; ++kt) {
            u32x8 a, bb;
#pragma unroll
            for (int r = 0; r < 8; ++r) {
                a[r]  = *(const unsigned int*)&q[qrow + kt * 32 + akb(r, half)];
                bb[r] = *(const unsigned int*)&k[((size_t)b * NPOS + j0 + 16 + l15) * CCH +
                                                 kt * 32 + bkb(r, half)];
            }
            s1 = wmma_bf16(a, bb, s1);
        }
        // online softmax update (row = r + 8*half, shared max over both tiles)
        float fac[8];
        bool need = false;
#pragma unroll
        for (int r = 0; r < 8; ++r) {
            float v0 = s0[r] * sc, v1 = s1[r] * sc;
            float mx = fmaxf(v0, v1);
            for (int off = 1; off < 16; off <<= 1)
                mx = fmaxf(mx, __shfl_xor(mx, off, 32));
            float mnew = fmaxf(m8[r], mx);
            float e0 = __expf(v0 - mnew);
            float e1 = __expf(v1 - mnew);
            float sum = e0 + e1;
            for (int off = 1; off < 16; off <<= 1)
                sum += __shfl_xor(sum, off, 32);
            fac[r] = __expf(m8[r] - mnew);
            need  |= (mnew != m8[r]);
            l8[r]  = l8[r] * fac[r] + sum;
            m8[r]  = mnew;
            pls[wave][(r + (half << 3)) * 34 + l15]      = f2bf(e0);
            pls[wave][(r + (half << 3)) * 34 + 16 + l15] = f2bf(e1);
        }
        if (__any(need)) {      // wave-uniform; rarely taken once max stabilizes
#pragma unroll
            for (int ct = 0; ct < 16; ++ct)
#pragma unroll
                for (int r = 0; r < 8; ++r) o[ct][r] *= fac[r];
        }
        __builtin_amdgcn_wave_barrier();   // DS ops are in-order per wave
        u32x8 pa;
#pragma unroll
        for (int r = 0; r < 8; ++r)
            pa[r] = *(const unsigned int*)&pls[wave][l15 * 34 + akb(r, half)];
        __builtin_amdgcn_wave_barrier();
        // O[ct] += P @ V  over 16 channel tiles (V stored (ch, pos), j contiguous)
        for (int ct = 0; ct < 16; ++ct) {
            u32x8 vb;
#pragma unroll
            for (int r = 0; r < 8; ++r)
                vb[r] = *(const unsigned int*)&v[((size_t)b * CCH + ct * 16 + l15) * NPOS +
                                                 j0 + bkb(r, half)];
            o[ct] = wmma_bf16(pa, vb, o[ct]);
        }
    }
    // normalize by l and store (pos, ch) bf16
    float rl[8];
#pragma unroll
    for (int r = 0; r < 8; ++r) rl[r] = 1.0f / l8[r];
    for (int ct = 0; ct < 16; ++ct) {
#pragma unroll
        for (int r = 0; r < 8; ++r) {
            int row = q0 + r + (half << 3);
            hat[((size_t)b * NPOS + row) * CCH + ct * 16 + l15] = f2bf(o[ct][r] * rl[r]);
        }
    }
}

// ---------------- Kernel 6: output projection + residual -----------------
__global__ __launch_bounds__(256) void proj_residual(const unsigned short* __restrict__ hat,
                                                     const unsigned short* __restrict__ wp,
                                                     const float* __restrict__ bp,
                                                     const float* __restrict__ x,
                                                     float* __restrict__ out) {
    int b  = blockIdx.y;
    int n0 = blockIdx.x * 64;
    __shared__ unsigned short sh[64 * 264];
    for (int i = threadIdx.x; i < 64 * 64; i += 256) {
        int row = i >> 6, c4 = (i & 63) * 4;
        *(uint2*)&sh[row * 264 + c4] =
            *(const uint2*)&hat[((size_t)(b * NPOS + n0 + row)) * CCH + c4];
    }
    __syncthreads();
    int lane = threadIdx.x & 31, wave = threadIdx.x >> 5;
    int half = lane >> 4, l15 = lane & 15;
    for (int t = wave * 8; t < wave * 8 + 8; ++t) {
        int mt = t >> 2, nt = t & 3;
        int o0 = mt * 16, prow = nt * 16;
        f32x8 acc = {0.f, 0.f, 0.f, 0.f, 0.f, 0.f, 0.f, 0.f};
        for (int kt = 0; kt < 8; ++kt) {
            int k0 = kt * 32;
            u32x8 a, bb;
#pragma unroll
            for (int r = 0; r < 8; ++r) {
                a[r]  = *(const unsigned int*)&wp[(o0 + l15) * CCH + k0 + akb(r, half)];
                bb[r] = *(const unsigned int*)&sh[(prow + l15) * 264 + k0 + bkb(r, half)];
            }
            acc = wmma_bf16(a, bb, acc);
        }
#pragma unroll
        for (int r = 0; r < 8; ++r) {
            int o   = o0 + r + (half << 3);
            int pos = n0 + prow + l15;         // lanes -> consecutive pixels: coalesced
            int f = pos / HWSZ, p = pos % HWSZ;
            size_t xi = ((size_t)(b * 3 + f) * CCH + o) * HWSZ + p;
            out[xi] = x[xi] + acc[r] + bp[o];
        }
    }
}

// -------------------------------------------------------------------------
extern "C" void kernel_launch(void* const* d_in, const int* in_sizes, int n_in,
                              void* d_out, int out_size, void* d_ws, size_t ws_size,
                              hipStream_t stream) {
    const float* x   = (const float*)d_in[0];
    const float* gsc = (const float*)d_in[1];
    const float* gbi = (const float*)d_in[2];
    const float* wq  = (const float*)d_in[3];
    const float* bq  = (const float*)d_in[4];
    const float* wk  = (const float*)d_in[5];
    const float* bk  = (const float*)d_in[6];
    const float* wv  = (const float*)d_in[7];
    const float* bv  = (const float*)d_in[8];
    const float* wp  = (const float*)d_in[9];
    const float* bp  = (const float*)d_in[10];
    float* out = (float*)d_out;
    (void)in_sizes; (void)n_in; (void)out_size; (void)ws_size;

    char* w = (char*)d_ws;
    float* meanrstd = (float*)w;                 w += 256;
    const size_t tsz = (size_t)BATCH * NPOS * CCH * sizeof(unsigned short); // 7077888
    unsigned short* hn  = (unsigned short*)w;    w += tsz;
    unsigned short* qb  = (unsigned short*)w;    w += tsz;
    unsigned short* kb  = (unsigned short*)w;    w += tsz;
    unsigned short* vb  = (unsigned short*)w;    w += tsz;
    unsigned short* hat = (unsigned short*)w;    w += tsz;
    const size_t wsz = (size_t)CCH * CCH * sizeof(unsigned short);          // 131072
    unsigned short* wqb = (unsigned short*)w;    w += wsz;
    unsigned short* wkb = (unsigned short*)w;    w += wsz;
    unsigned short* wvb = (unsigned short*)w;    w += wsz;
    unsigned short* wpb = (unsigned short*)w;    w += wsz;

    gn_stats<<<BATCH * NGRP, 256, 0, stream>>>(x, meanrstd);
    gn_apply<<<dim3(HWSZ / 64, CCH / 64, BATCH * FF), 256, 0, stream>>>(x, meanrstd, gsc, gbi, hn);

    convert_w<<<256, 256, 0, stream>>>(wq, wqb, CCH * CCH);
    convert_w<<<256, 256, 0, stream>>>(wk, wkb, CCH * CCH);
    convert_w<<<256, 256, 0, stream>>>(wv, wvb, CCH * CCH);
    convert_w<<<256, 256, 0, stream>>>(wp, wpb, CCH * CCH);

    dim3 g64(NPOS / 64, BATCH);
    qkv_gemm<<<g64, 256, 0, stream>>>(hn, wqb, bq, qb, 0);
    qkv_gemm<<<g64, 256, 0, stream>>>(hn, wkb, bk, kb, 0);
    qkv_gemm<<<g64, 256, 0, stream>>>(hn, wvb, bv, vb, 1);

    dim3 g128(NPOS / 128, BATCH);
    flash_attn<<<g128, 256, 0, stream>>>(qb, kb, vb, hat);

    proj_residual<<<g64, 256, 0, stream>>>(hat, wpb, bp, x, out);
}
